// DispEstimator_3238405341262
// MI455X (gfx1250) — matrix operand order repaired and checked by reading, not compiled, not run on hardware
//
#include <hip/hip_runtime.h>

// -------- fixed problem geometry (from reference) --------
#define HH   256
#define WWID 256
#define HWSZ 65536   // 256*256
#define BATCH 4

typedef __attribute__((ext_vector_type(16))) __bf16 v16bf;
typedef __attribute__((ext_vector_type(8)))  float  v8f;

__constant__ float GW[7] = {0.03663285f, 0.11128076f, 0.21674532f,
                            0.27068215f, 0.21674532f, 0.11128076f, 0.03663285f};

__device__ __forceinline__ float nan0f(float v) {
    return __builtin_isfinite(v) ? v : 0.0f;
}
__device__ __forceinline__ int clampi(int v, int lo, int hi) {
    return v < lo ? lo : (v > hi ? hi : v);
}

// =====================================================================
// Weight repack: W[co][ci][r] (f32) -> Wb[co][r*Cinp + ci] (bf16),
// zero-padded to co < Coutp, ci < Cinp.  K-order r-major so that one
// 32-wide K-step has a uniform kernel tap r.
// =====================================================================
__global__ __launch_bounds__(256)
void repack_w(const float* __restrict__ W, __bf16* __restrict__ Wb,
              int Cout, int Coutp, int Cin, int Cinp)
{
    const int K = 9 * Cinp;
    const int idx = blockIdx.x * 256 + threadIdx.x;
    if (idx >= Coutp * K) return;
    const int co = idx / K;
    const int k  = idx - co * K;
    const int r  = k / Cinp;
    const int ci = k - r * Cinp;
    float v = 0.0f;
    if (co < Cout && ci < Cin) v = W[((size_t)co * Cin + ci) * 9 + r];
    Wb[idx] = (__bf16)v;
}

// =====================================================================
// Implicit-GEMM 3x3 conv ("same", pad == dilation), bf16 WMMA, f32 acc.
// Block: 256 thr = 8 wave32 (2 cout x 4 pixel groups).
// Tile: 64 Cout x 64 pixels; each wave: 32x16 (2 WMMAs per K-step of 32).
// Cin padded to power-of-two Cinp (pad channels of input must be zero).
// =====================================================================
__global__ __launch_bounds__(256)
void conv3x3_wmma(const float* __restrict__ In, int lgCin, int in_bstride,
                  const __bf16* __restrict__ Wb, const float* __restrict__ Bias,
                  float* __restrict__ Out, int Cout, int out_bstride, int dil)
{
    const int Cinp = 1 << lgCin;
    const int Kpad = 9 << lgCin;

    __shared__ __bf16 Asm[64][36];   // [cout_local][k], +4 skew (72B rows)
    __shared__ __bf16 Bsm[64][36];   // [pixel_local][k]

    const int tid   = threadIdx.x;
    const int b     = blockIdx.z;
    const int cout0 = blockIdx.y << 6;
    const int y     = blockIdx.x >> 2;           // 256 rows
    const int x0    = (blockIdx.x & 3) << 6;     // 4 segments of 64

    const float* __restrict__ inB = In + (size_t)b * in_bstride;

    // staging map: 64 rows x (4 groups of 8 k)
    const int sn  = tid & 63;
    const int skb = (tid >> 6) << 3;

    // wave / fragment maps (wave32)
    const int wave = tid >> 5, lane = tid & 31;
    const int wr = wave >> 2, wc = wave & 3;
    const int hi = lane >> 4, lo = lane & 15;

    v8f acc0 = {}, acc1 = {};

    const uint4* __restrict__ Wv =
        (const uint4*)(Wb + (size_t)(cout0 + sn) * Kpad);

    for (int k0 = 0; k0 < Kpad; k0 += 32) {
        // geometry for this K-step: kernel tap r is uniform (Cinp >= 32)
        const int r   = k0 >> lgCin;
        const int ci0 = k0 & (Cinp - 1);
        const int yy  = y + (r / 3 - 1) * dil;              // scalar
        const int xx  = x0 + sn + (r % 3 - 1) * dil;        // per thread
        const bool inb = ((unsigned)yy < HH) && ((unsigned)xx < WWID);

        __syncthreads();   // WAR on previous iteration's fragment reads

        // ---- stage A: 8 bf16 from packed weights (one uint4 load) ----
        {
            const uint4 w4 = Wv[(k0 + skb) >> 3];
            uint2 t0; t0.x = w4.x; t0.y = w4.y;
            uint2 t1; t1.x = w4.z; t1.y = w4.w;
            *(uint2*)(&Asm[sn][skb])     = t0;   // 8B-aligned (72B rows)
            *(uint2*)(&Asm[sn][skb + 4]) = t1;
        }

        // ---- stage B: 8 unconditional-in-branch coalesced loads ----
        {
            float f[8];
            if (inb) {
                const float* src =
                    inB + (size_t)(ci0 + skb) * HWSZ + yy * WWID + xx;
                #pragma unroll
                for (int j = 0; j < 8; ++j) f[j] = src[(size_t)j * HWSZ];
            } else {
                #pragma unroll
                for (int j = 0; j < 8; ++j) f[j] = 0.0f;
            }
            alignas(16) __bf16 h[8];
            #pragma unroll
            for (int j = 0; j < 8; ++j) h[j] = (__bf16)f[j];
            *(uint2*)(&Bsm[sn][skb])     = *(const uint2*)(&h[0]);
            *(uint2*)(&Bsm[sn][skb + 4]) = *(const uint2*)(&h[4]);
        }
        __syncthreads();

        // ---- fragments (ISA 7.12.2 bf16 layouts) + 2x WMMA ----
        union { v16bf v; unsigned u[8]; } a0, a1, bb;
        const unsigned* Ar0 = (const unsigned*)(&Asm[wr * 32 + lo][0]);
        const unsigned* Ar1 = (const unsigned*)(&Asm[wr * 32 + 16 + lo][0]);
        const unsigned* Br  = (const unsigned*)(&Bsm[wc * 16 + lo][0]);
        #pragma unroll
        for (int v2 = 0; v2 < 8; ++v2) {
            const int ka = ((v2 < 4) ? (2 * v2) : (2 * v2 + 8)) + hi * 8;  // A 16x32
            const int kb = ((v2 < 4) ? (2 * v2) : (2 * v2 + 8)) + hi * 8;  // B 32x16
            a0.u[v2] = Ar0[ka >> 1];
            a1.u[v2] = Ar1[ka >> 1];
            bb.u[v2] = Br[kb >> 1];
        }
        acc0 = __builtin_amdgcn_wmma_f32_16x16x32_bf16(
                   false, a0.v, false, bb.v, (short)0, acc0, false, false);
        acc1 = __builtin_amdgcn_wmma_f32_16x16x32_bf16(
                   false, a1.v, false, bb.v, (short)0, acc1, false, false);
    }

    // ---- epilogue: D layout n=lo, m = r + hi*8 ----
    const int x = x0 + wc * 16 + lo;
    float* outB = Out + (size_t)b * out_bstride;
    #pragma unroll
    for (int s = 0; s < 2; ++s) {
        const v8f a = s ? acc1 : acc0;
        #pragma unroll
        for (int rr = 0; rr < 8; ++rr) {
            const int co = cout0 + wr * 32 + s * 16 + hi * 8 + rr;
            if (co < Cout) {
                float o = a[rr];
                if (Bias) o += Bias[co];
                outB[(size_t)co * HWSZ + y * WWID + x] = o;
            }
        }
    }
}

// =====================================================================
// Instance norm: per-(b,c) mean & rsqrt(var+eps)
// =====================================================================
__global__ __launch_bounds__(256)
void in_stats(const float* __restrict__ X, float* __restrict__ Mean,
              float* __restrict__ Rsig)
{
    const int bc = blockIdx.x;
    const float* p = X + (size_t)bc * HWSZ;
    float s = 0.f, s2 = 0.f;
    for (int i = threadIdx.x; i < HWSZ; i += 256) {
        float v = p[i]; s += v; s2 += v * v;
    }
    __shared__ float r1[256], r2[256];
    r1[threadIdx.x] = s; r2[threadIdx.x] = s2;
    __syncthreads();
    for (int st = 128; st > 0; st >>= 1) {
        if (threadIdx.x < st) {
            r1[threadIdx.x] += r1[threadIdx.x + st];
            r2[threadIdx.x] += r2[threadIdx.x + st];
        }
        __syncthreads();
    }
    if (threadIdx.x == 0) {
        float mu  = r1[0] * (1.0f / HWSZ);
        float var = r2[0] * (1.0f / HWSZ) - mu * mu;
        Mean[bc] = mu;
        Rsig[bc] = rsqrtf(var + 1e-5f);
    }
}

__global__ __launch_bounds__(256)
void in_apply(float* __restrict__ X, const float* __restrict__ Mean,
              const float* __restrict__ Rsig, const float* __restrict__ G,
              const float* __restrict__ Be, int C)
{
    const int bc = blockIdx.y;
    const int c  = bc % C;
    float* p = X + (size_t)bc * HWSZ;
    const int i = blockIdx.x * 256 + threadIdx.x;
    float v = (p[i] - Mean[bc]) * Rsig[bc] * G[c] + Be[c];
    p[i] = v >= 0.f ? v : 0.2f * v;
}

// =====================================================================
// Depthwise 7x7 Gaussian blur, edge-replicate padding. grid (256, C, B)
// =====================================================================
__global__ __launch_bounds__(256)
void gauss7(const float* __restrict__ In, float* __restrict__ Out,
            int in_bstride, int out_bstride, int nanz)
{
    const int b = blockIdx.z, c = blockIdx.y;
    const int p = blockIdx.x * 256 + threadIdx.x;
    const int y = p >> 8, x = p & 255;
    const float* src = In + (size_t)b * in_bstride + (size_t)c * HWSZ;
    float acc = 0.f;
    #pragma unroll
    for (int dy = 0; dy < 7; ++dy) {
        const int yy = clampi(y + dy - 3, 0, 255);
        const float wy = GW[dy];
        #pragma unroll
        for (int dx = 0; dx < 7; ++dx) {
            const int xx = clampi(x + dx - 3, 0, 255);
            acc += wy * GW[dx] * src[yy * WWID + xx];
        }
    }
    if (nanz) acc = nan0f(acc);
    Out[(size_t)b * out_bstride + (size_t)c * HWSZ + p] = acc;
}

// =====================================================================
// Local correlation: lc[b,q,y,x] = mean_c (f2 - shifted(blur f1))^2
// f2 = Cat channels 64..127; writes E channels 64..112 (E has 128 ch).
// Unconditional clamped loads + select (zero pad semantics).
// =====================================================================
__global__ __launch_bounds__(256)
void corr49(const float* __restrict__ Cat, const float* __restrict__ F1s,
            float* __restrict__ E, float* __restrict__ Acc)
{
    const int b = blockIdx.y;
    const int p = blockIdx.x * 256 + threadIdx.x;
    const int y = p >> 8, x = p & 255;
    const float* F2 = Cat + (size_t)b * 128 * HWSZ + (size_t)64 * HWSZ;
    const float* F1 = F1s + (size_t)b * 64 * HWSZ;
    float a[49];
    #pragma unroll
    for (int q = 0; q < 49; ++q) a[q] = 0.f;
    for (int c = 0; c < 64; ++c) {
        const float v2 = F2[(size_t)c * HWSZ + p];
        const float* base = F1 + (size_t)c * HWSZ;
        #pragma unroll
        for (int q = 0; q < 49; ++q) {
            const int yy = y + q / 7 - 3;
            const int xx = x + q % 7 - 3;
            const int yc = clampi(yy, 0, 255), xc = clampi(xx, 0, 255);
            const float ld = base[yc * WWID + xc];
            const bool ib = ((unsigned)yy < HH) && ((unsigned)xx < WWID);
            const float v1 = ib ? ld : 0.f;
            const float d = v2 - v1;
            a[q] += d * d;
        }
    }
    float lsum = 0.f;
    float* Eb = E + (size_t)b * 128 * HWSZ + (size_t)64 * HWSZ;
    #pragma unroll
    for (int q = 0; q < 49; ++q) {
        const float m = a[q] * (1.0f / 64.0f);
        Eb[(size_t)q * HWSZ + p] = m;
        lsum += m;
    }
    __shared__ float red[256];
    red[threadIdx.x] = lsum;
    __syncthreads();
    for (int st = 128; st > 0; st >>= 1) {
        if (threadIdx.x < st) red[threadIdx.x] += red[threadIdx.x + st];
        __syncthreads();
    }
    if (threadIdx.x == 0) atomicAdd(&Acc[b], red[0]);
}

// normalize lc by per-b mean, nan0 everything, zero pad channels.
// grid (256, 128, B): c<64 nan0; 64<=c<113 scale+nan0; c>=113 write 0.
__global__ __launch_bounds__(256)
void corr_finalize(float* __restrict__ E, const float* __restrict__ Acc)
{
    const int b = blockIdx.z, c = blockIdx.y;
    const int p = blockIdx.x * 256 + threadIdx.x;
    float* ptr = E + ((size_t)b * 128 + c) * HWSZ + p;
    float v;
    if (c >= 113) {
        v = 0.0f;
    } else {
        v = *ptr;
        if (c >= 64) {
            const float mean = Acc[b] * (1.0f / (49.0f * (float)HWSZ));
            v = v / (mean + 1e-6f);
        }
        v = nan0f(v);
    }
    *ptr = v;
}

// tiny head conv 16 -> 2, pad 1, zero pad. grid (256, 2, B)
__global__ __launch_bounds__(256)
void head_conv(const float* __restrict__ X, const float* __restrict__ Wt,
               const float* __restrict__ Bias, float* __restrict__ T)
{
    const int b = blockIdx.z, co = blockIdx.y;
    const int p = blockIdx.x * 256 + threadIdx.x;
    const int y = p >> 8, x = p & 255;
    const float* xb = X + (size_t)b * 16 * HWSZ;
    float s = Bias[co];
    for (int ci = 0; ci < 16; ++ci) {
        const float* src = xb + (size_t)ci * HWSZ;
        const float* w = Wt + ((size_t)co * 16 + ci) * 9;
        #pragma unroll
        for (int r = 0; r < 9; ++r) {
            const int yy = y + r / 3 - 1, xx = x + r % 3 - 1;
            const int yc = clampi(yy, 0, 255), xc = clampi(xx, 0, 255);
            const float ld = src[yc * WWID + xc];
            const bool ib = ((unsigned)yy < HH) && ((unsigned)xx < WWID);
            s += w[r] * (ib ? ld : 0.f);
        }
    }
    T[((size_t)b * 2 + co) * HWSZ + p] = s;
}

// =====================================================================
extern "C" void kernel_launch(void* const* d_in, const int* in_sizes, int n_in,
                              void* d_out, int out_size, void* d_ws, size_t ws_size,
                              hipStream_t stream)
{
    const float* feat1  = (const float*)d_in[0];
    const float* feat2  = (const float*)d_in[1];
    const float* pre_w  = (const float*)d_in[2];
    const float* pre_b  = (const float*)d_in[3];
    const float* fc1_w  = (const float*)d_in[4];
    const float* fc1_g  = (const float*)d_in[5];
    const float* fc1_be = (const float*)d_in[6];
    const float* fc2_w  = (const float*)d_in[7];
    const float* fc2_b  = (const float*)d_in[8];
    const float* e1_w   = (const float*)d_in[9];
    const float* e1_g   = (const float*)d_in[10];
    const float* e1_be  = (const float*)d_in[11];
    const float* e2_w   = (const float*)d_in[12];
    const float* e2_g   = (const float*)d_in[13];
    const float* e2_be  = (const float*)d_in[14];
    const float* e3_w   = (const float*)d_in[15];
    const float* e3_g   = (const float*)d_in[16];
    const float* e3_be  = (const float*)d_in[17];
    const float* head_w = (const float*)d_in[18];
    const float* head_b = (const float*)d_in[19];
    float* out = (float*)d_out;

    const size_t HW = HWSZ;
    float* ws   = (float*)d_ws;
    float* CAT  = ws;                        // [4,128,HW] pre outputs (f1|f2)
    float* FC1  = ws + 4ull * 128 * HW;      // [4,128,HW]
    float* E    = ws + 8ull * 128 * HW;      // [4,128,HW] feat|lc|zero-pad
    float* F1S  = ws + 12ull * 128 * HW;     // [4,64,HW] blurred f1
    float* F    = CAT;                       // reuse: e1 out [4,64,HW]
    float* G    = CAT + 4ull * 64 * HW;      // e2 out [4,32,HW]
    float* Hb   = G + 4ull * 32 * HW;        // e3 out [4,16,HW]
    float* T    = Hb + 4ull * 16 * HW;       // head out [4,2,HW]
    float* MEAN = F1S + 4ull * 64 * HW;      // 1024
    float* RSIG = MEAN + 1024;               // 1024
    float* ACC  = RSIG + 1024;               // 4 used

    // packed bf16 weights
    __bf16* WBpre = (__bf16*)(ACC + 1024);
    __bf16* WBfc1 = WBpre + 64ull * 576;     //  36864
    __bf16* WBfc2 = WBfc1 + 128ull * 1152;   // 147456
    __bf16* WBe1  = WBfc2 + 64ull * 1152;    //  73728
    __bf16* WBe2  = WBe1  + 64ull * 1152;    //  73728
    __bf16* WBe3  = WBe2  + 64ull * 576;     //  36864

    (void)hipMemsetAsync(ACC, 0, 4 * sizeof(float), stream);

    dim3 blk(256);

    // ---- weight repacking (tiny) ----
    repack_w<<<dim3((64 * 576 + 255) / 256), blk, 0, stream>>>(pre_w, WBpre, 64, 64, 64, 64);
    repack_w<<<dim3((128 * 1152 + 255) / 256), blk, 0, stream>>>(fc1_w, WBfc1, 128, 128, 128, 128);
    repack_w<<<dim3((64 * 1152 + 255) / 256), blk, 0, stream>>>(fc2_w, WBfc2, 64, 64, 128, 128);
    repack_w<<<dim3((64 * 1152 + 255) / 256), blk, 0, stream>>>(e1_w, WBe1, 64, 64, 113, 128);
    repack_w<<<dim3((64 * 576 + 255) / 256), blk, 0, stream>>>(e2_w, WBe2, 32, 64, 64, 64);
    repack_w<<<dim3((64 * 288 + 255) / 256), blk, 0, stream>>>(e3_w, WBe3, 16, 64, 32, 32);

    // 1) preprocessor on both features -> CAT halves
    conv3x3_wmma<<<dim3(1024, 1, BATCH), blk, 0, stream>>>(
        feat1, 6, 64 * HWSZ, WBpre, pre_b, CAT, 64, 128 * HWSZ, 1);
    conv3x3_wmma<<<dim3(1024, 1, BATCH), blk, 0, stream>>>(
        feat2, 6, 64 * HWSZ, WBpre, pre_b, CAT + 64 * HW, 64, 128 * HWSZ, 1);

    // 2) featcompressor layer1 + IN + LReLU
    conv3x3_wmma<<<dim3(1024, 2, BATCH), blk, 0, stream>>>(
        CAT, 7, 128 * HWSZ, WBfc1, nullptr, FC1, 128, 128 * HWSZ, 1);
    in_stats<<<dim3(BATCH * 128), blk, 0, stream>>>(FC1, MEAN, RSIG);
    in_apply<<<dim3(256, BATCH * 128), blk, 0, stream>>>(FC1, MEAN, RSIG, fc1_g, fc1_be, 128);

    // 3) featcompressor layer2 -> E channels 0..63
    conv3x3_wmma<<<dim3(1024, 1, BATCH), blk, 0, stream>>>(
        FC1, 7, 128 * HWSZ, WBfc2, fc2_b, E, 64, 128 * HWSZ, 1);

    // 4) gaussian blur of f1 (CAT channels 0..63)
    gauss7<<<dim3(256, 64, BATCH), blk, 0, stream>>>(CAT, F1S, 128 * HWSZ, 64 * HWSZ, 0);

    // 5) local correlation -> E channels 64..112, then normalize + nan0 + pad
    corr49<<<dim3(256, BATCH), blk, 0, stream>>>(CAT, F1S, E, ACC);
    corr_finalize<<<dim3(256, 128, BATCH), blk, 0, stream>>>(E, ACC);

    // 6) estimator stack (dilated convs + IN + LReLU)
    conv3x3_wmma<<<dim3(1024, 1, BATCH), blk, 0, stream>>>(
        E, 7, 128 * HWSZ, WBe1, nullptr, F, 64, 64 * HWSZ, 1);
    in_stats<<<dim3(BATCH * 64), blk, 0, stream>>>(F, MEAN, RSIG);
    in_apply<<<dim3(256, BATCH * 64), blk, 0, stream>>>(F, MEAN, RSIG, e1_g, e1_be, 64);

    conv3x3_wmma<<<dim3(1024, 1, BATCH), blk, 0, stream>>>(
        F, 6, 64 * HWSZ, WBe2, nullptr, G, 32, 32 * HWSZ, 2);
    in_stats<<<dim3(BATCH * 32), blk, 0, stream>>>(G, MEAN, RSIG);
    in_apply<<<dim3(256, BATCH * 32), blk, 0, stream>>>(G, MEAN, RSIG, e2_g, e2_be, 32);

    conv3x3_wmma<<<dim3(1024, 1, BATCH), blk, 0, stream>>>(
        G, 5, 32 * HWSZ, WBe3, nullptr, Hb, 16, 16 * HWSZ, 4);
    in_stats<<<dim3(BATCH * 16), blk, 0, stream>>>(Hb, MEAN, RSIG);
    in_apply<<<dim3(256, BATCH * 16), blk, 0, stream>>>(Hb, MEAN, RSIG, e3_g, e3_be, 16);

    // 7) flow head + final blur (+nan0) into d_out
    head_conv<<<dim3(256, 2, BATCH), blk, 0, stream>>>(Hb, head_w, head_b, T);
    gauss7<<<dim3(256, 2, BATCH), blk, 0, stream>>>(T, out, 2 * HWSZ, 2 * HWSZ, 1);
}
